// TKLinearAttention_67628555042885
// MI455X (gfx1250) — matrix operand order repaired and checked by reading, not compiled, not run on hardware
//
#include <hip/hip_runtime.h>
#include <hip/hip_bf16.h>

// ---------------------------------------------------------------------------
// Linear attention (elu+1 feature map), B=4 N=4096 D=1024 H=16 hd=64.
// bf16 WMMA everywhere a matmul appears; async global->LDS staging (ASYNCcnt)
// with double buffering for the big GEMMs.
// ---------------------------------------------------------------------------

#define D_MODEL 1024
#define N_HEADS 16
#define HEAD_DIM 64
#define BSZ 4
#define NSEQ 4096
#define MROWS (BSZ * NSEQ)          // 16384
#define EPS 1e-6f
#define KVSPLIT 8

typedef __attribute__((ext_vector_type(16))) __bf16 v16bf;
typedef __attribute__((ext_vector_type(8)))  __bf16 v8bf;
typedef __attribute__((ext_vector_type(8)))  float  v8f;

union V16U { v16bf v; v8bf h[2]; };

__device__ __forceinline__ float feat_map(float x) {
    return x > 0.0f ? x + 1.0f : __expf(x);
}

// Async 16-byte global -> LDS copy (per-lane addresses), tracked by ASYNCcnt.
__device__ __forceinline__ void async_copy_b128(unsigned lds_off, const void* gaddr) {
    asm volatile("global_load_async_to_lds_b128 %0, %1, off"
                 :: "v"(lds_off), "v"(gaddr) : "memory");
}

__device__ __forceinline__ void wait_async0() {
#if defined(__has_builtin)
#if __has_builtin(__builtin_amdgcn_s_wait_asynccnt)
    __builtin_amdgcn_s_wait_asynccnt(0);
#else
    asm volatile("s_wait_asynccnt 0x0" ::: "memory");
#endif
#else
    asm volatile("s_wait_asynccnt 0x0" ::: "memory");
#endif
}

// ---------------------------------------------------------------------------
// fp32 -> bf16 conversion (grid-stride)
// ---------------------------------------------------------------------------
__global__ __launch_bounds__(256)
void f2bf_kernel(const float* __restrict__ src, __bf16* __restrict__ dst, int n) {
    int i = blockIdx.x * blockDim.x + threadIdx.x;
    int stride = gridDim.x * blockDim.x;
    for (; i < n; i += stride) dst[i] = (__bf16)src[i];
}

__global__ __launch_bounds__(256)
void zero_kernel(float* __restrict__ dst, int n) {
    int i = blockIdx.x * blockDim.x + threadIdx.x;
    int stride = gridDim.x * blockDim.x;
    for (; i < n; i += stride) dst[i] = 0.0f;
}

// ---------------------------------------------------------------------------
// GEMM: Y = A[M,K] @ W[Nd,K]^T + bias, bf16 in, fp32 WMMA accum.
// Wave tile: 32 rows x 64 cols (2 A operands x 4 B operands -> 8 WMMAs/step).
// Block: 8 waves -> 256 rows x 64 cols. W panel (64x32 per k-step) staged in
// LDS via async global->LDS copies, double buffered, one barrier per k-step.
// ---------------------------------------------------------------------------
template<bool PERM, bool F32OUT>
__global__ __launch_bounds__(256)
void gemm_bf16_kernel(const __bf16* __restrict__ A,
                      const __bf16* __restrict__ W,
                      const float*  __restrict__ bias,
                      void* __restrict__ Y,
                      int featmap)
{
    constexpr int BSTR = 40;   // padded halfs per W-tile row (80B, bank-friendly)
    __shared__ __align__(16) __bf16 wtile[2][64 * BSTR];

    const int lane  = threadIdx.x & 31;
    const int wave  = threadIdx.x >> 5;
    const int l16   = lane & 15;
    const int hi    = lane >> 4;             // 0 or 1
    const int mbase = blockIdx.y * 256 + wave * 32;
    const int nbase = blockIdx.x * 64;

    // cooperative staging: thread t copies 16B of the 64x32 W panel
    const int srow = threadIdx.x >> 2;        // 0..63 (W row within panel)
    const int scol = (threadIdx.x & 3) * 8;   // half offset 0/8/16/24
    const __bf16* wsrc = W + (size_t)(nbase + srow) * D_MODEL + scol;
    const unsigned lds0 = (unsigned)(uintptr_t)(&wtile[0][srow * BSTR + scol]);
    const unsigned lds1 = (unsigned)(uintptr_t)(&wtile[1][srow * BSTR + scol]);

    const __bf16* arow0 = A + (size_t)(mbase + l16) * D_MODEL;
    const __bf16* arow1 = arow0 + (size_t)16 * D_MODEL;

    v8f c0[4] = {v8f{}, v8f{}, v8f{}, v8f{}};
    v8f c1[4] = {v8f{}, v8f{}, v8f{}, v8f{}};

    constexpr int NITER = D_MODEL / 32;
    async_copy_b128(lds0, wsrc);              // prefetch k-step 0 into buf 0

    for (int it = 0; it < NITER; ++it) {
        const int kk = it * 32;
        wait_async0();                        // my slice of current panel done
        __syncthreads();                      // everyone's slice visible; prior
                                              // reads of the other buffer done
        if (it + 1 < NITER)                   // prefetch next panel
            async_copy_b128((it & 1) ? lds0 : lds1, wsrc + kk + 32);

        V16U a0, a1;
        a0.h[0] = *(const v8bf*)(arow0 + kk + hi * 8);
        a0.h[1] = *(const v8bf*)(arow0 + kk + 16 + hi * 8);
        a1.h[0] = *(const v8bf*)(arow1 + kk + hi * 8);
        a1.h[1] = *(const v8bf*)(arow1 + kk + 16 + hi * 8);

        const __bf16* bt = wtile[it & 1];
#pragma unroll
        for (int j = 0; j < 4; ++j) {
            const __bf16* bp = bt + (j * 16 + l16) * BSTR + hi * 8;
            V16U b;
            b.h[0] = *(const v8bf*)(bp);
            b.h[1] = *(const v8bf*)(bp + 16);
            c0[j] = __builtin_amdgcn_wmma_f32_16x16x32_bf16(
                false, a0.v, false, b.v, (short)0, c0[j], false, false);
            c1[j] = __builtin_amdgcn_wmma_f32_16x16x32_bf16(
                false, a1.v, false, b.v, (short)0, c1[j], false, false);
        }
    }

#pragma unroll
    for (int j = 0; j < 4; ++j) {
        const int ncol = nbase + j * 16 + l16;
        const float bv = bias[ncol];
#pragma unroll
        for (int rt = 0; rt < 16; ++rt) {     // 2 row tiles x 8 regs
            const int r    = rt & 7;
            const int mrow = mbase + (rt >> 3) * 16 + r + hi * 8;
            float val = ((rt < 8) ? c0[j][r] : c1[j][r]) + bv;
            if (featmap) val = feat_map(val);
            if (F32OUT) {
                ((float*)Y)[(size_t)mrow * D_MODEL + ncol] = val;
            } else if (PERM) {
                const int b  = mrow >> 12;            // / NSEQ
                const int n  = mrow & (NSEQ - 1);
                const int h  = ncol >> 6;
                const int dd = ncol & 63;
                ((__bf16*)Y)[(((size_t)(b * N_HEADS + h) * NSEQ + n) << 6) + dd] =
                    (__bf16)val;
            } else {
                ((__bf16*)Y)[(size_t)mrow * D_MODEL + ncol] = (__bf16)val;
            }
        }
    }
}

// ---------------------------------------------------------------------------
// Per-head kv state via WMMA: kvAcc[m][d] += sum_n k[n,d] * v[n,m] (fp32
// atomics across KVSPLIT N-slices), ksAcc[d] += sum_n k[n,d].
// K,V tiles are staged TRANSPOSED into LDS (scatter b16 stores) so both WMMA
// operands read contiguous halfs. One block per (b,h,slice); 8 waves each own
// a (16d x 32m) output strip sharing one A operand.
// ---------------------------------------------------------------------------
__global__ __launch_bounds__(256)
void kv_ksum_kernel(const __bf16* __restrict__ Kf,
                    const __bf16* __restrict__ Vf,
                    float* __restrict__ kvAcc,
                    float* __restrict__ ksAcc)
{
    constexpr int TSTR = 72;   // padded halfs per row (multiple of 8)
    __shared__ __align__(16) __bf16 Kt[64 * TSTR];   // [d][n]
    __shared__ __align__(16) __bf16 Vt[64 * TSTR];   // [m][n]

    const int bh    = blockIdx.x;
    const int slice = blockIdx.y;
    const __bf16* kh = Kf + ((size_t)bh * NSEQ + (size_t)slice * (NSEQ / KVSPLIT)) * HEAD_DIM;
    const __bf16* vh = Vf + ((size_t)bh * NSEQ + (size_t)slice * (NSEQ / KVSPLIT)) * HEAD_DIM;

    const int t    = threadIdx.x;
    const int lane = t & 31;
    const int wave = t >> 5;
    const int l16  = lane & 15;
    const int hi   = lane >> 4;
    const int di   = wave >> 1;          // d-tile 0..3
    const int mi0  = (wave & 1) * 2;     // first m-tile (0 or 2)

    const int srown = t >> 2;            // source row n (0..63)
    const int sd0   = (t & 3) * 16;      // 16-wide d/m block this thread scatters

    v8f c[2] = {v8f{}, v8f{}};
    float ks = 0.0f;

    for (int nc = 0; nc < NSEQ / KVSPLIT; nc += 64) {
        V16U kr, vr;
        const __bf16* krow = kh + (size_t)(nc + srown) * 64 + sd0;
        const __bf16* vrow = vh + (size_t)(nc + srown) * 64 + sd0;
        kr.h[0] = *(const v8bf*)(krow);
        kr.h[1] = *(const v8bf*)(krow + 8);
        vr.h[0] = *(const v8bf*)(vrow);
        vr.h[1] = *(const v8bf*)(vrow + 8);
        __syncthreads();                 // previous chunk fully consumed
#pragma unroll
        for (int i = 0; i < 16; ++i) {
            Kt[(sd0 + i) * TSTR + srown] = ((const __bf16*)&kr)[i];
            Vt[(sd0 + i) * TSTR + srown] = ((const __bf16*)&vr)[i];
        }
        __syncthreads();                 // transposed tiles visible

#pragma unroll
        for (int kc = 0; kc < 64; kc += 32) {
            V16U a;
            const __bf16* ap = Kt + (di * 16 + l16) * TSTR + kc + hi * 8;
            a.h[0] = *(const v8bf*)(ap);
            a.h[1] = *(const v8bf*)(ap + 16);
#pragma unroll
            for (int jm = 0; jm < 2; ++jm) {
                const __bf16* bp = Vt + ((mi0 + jm) * 16 + l16) * TSTR + kc + hi * 8;
                V16U b;
                b.h[0] = *(const v8bf*)(bp);
                b.h[1] = *(const v8bf*)(bp + 16);
                c[jm] = __builtin_amdgcn_wmma_f32_16x16x32_bf16(
                    false, a.v, false, b.v, (short)0, c[jm], false, false);
            }
        }
        if (t < 64) {                    // ksum partial for dim d = t
#pragma unroll 8
            for (int n = 0; n < 64; ++n) ks += (float)Kt[t * TSTR + n];
        }
    }

#pragma unroll
    for (int jm = 0; jm < 2; ++jm) {
        const int m = (mi0 + jm) * 16 + l16;
#pragma unroll
        for (int r = 0; r < 8; ++r) {
            const int d = di * 16 + r + hi * 8;
            unsafeAtomicAdd(&kvAcc[(size_t)bh * (HEAD_DIM * HEAD_DIM) + m * 64 + d],
                            c[jm][r]);
        }
    }
    if (t < 64) unsafeAtomicAdd(&ksAcc[bh * 64 + t], ks);
}

// ---------------------------------------------------------------------------
// Per-head: out[n,m] = (q[n,:] @ kvT[m,:]) / max(q[n,:].ksum, EPS)
// WMMA over K=64 (2 steps). Each wave: 16 rows x 64 cols (whole head width).
// ---------------------------------------------------------------------------
__global__ __launch_bounds__(256)
void qkv_norm_kernel(const __bf16* __restrict__ Qf,
                     const __bf16* __restrict__ kvT,
                     const float*  __restrict__ ksum,
                     __bf16* __restrict__ ctx)
{
    const int bh   = blockIdx.y;
    const int b    = bh >> 4;
    const int h    = bh & 15;
    const int lane = threadIdx.x & 31;
    const int wave = threadIdx.x >> 5;
    const int l16  = lane & 15;
    const int hi   = lane >> 4;
    const int rb   = blockIdx.x * 128 + wave * 16;

    const __bf16* qh  = Qf  + (size_t)bh * NSEQ * HEAD_DIM;
    const __bf16* kvh = kvT + (size_t)bh * (HEAD_DIM * HEAD_DIM);
    const float*  ksh = ksum + bh * 64;

    v8f c[4] = {v8f{}, v8f{}, v8f{}, v8f{}};

#pragma unroll
    for (int kk = 0; kk < HEAD_DIM; kk += 32) {
        V16U a;
        const __bf16* ar = qh + (size_t)(rb + l16) * 64 + kk + hi * 8;
        a.h[0] = *(const v8bf*)(ar);
        a.h[1] = *(const v8bf*)(ar + 16);
#pragma unroll
        for (int j = 0; j < 4; ++j) {
            const __bf16* br = kvh + (size_t)(j * 16 + l16) * 64 + kk + hi * 8;
            V16U bb;
            bb.h[0] = *(const v8bf*)(br);
            bb.h[1] = *(const v8bf*)(br + 16);
            c[j] = __builtin_amdgcn_wmma_f32_16x16x32_bf16(
                false, a.v, false, bb.v, (short)0, c[j], false, false);
        }
    }

    const __bf16* qr = qh + (size_t)(rb + l16) * 64 + hi * 32;
    float s = 0.0f;
#pragma unroll
    for (int dd = 0; dd < 32; ++dd)
        s = fmaf((float)qr[dd], ksh[hi * 32 + dd], s);
    s += __shfl_xor(s, 16, 32);
    s = fmaxf(s, EPS);

#pragma unroll
    for (int j = 0; j < 4; ++j) {
#pragma unroll
        for (int r = 0; r < 8; ++r) {
            const int rt   = r + hi * 8;
            const float nr = __shfl(s, rt, 32);
            const int n    = rb + rt;
            const int m    = j * 16 + l16;
            ctx[(((size_t)(b * NSEQ + n)) << 10) + h * 64 + m] =
                (__bf16)(c[j][r] / nr);
        }
    }
}

// ---------------------------------------------------------------------------
// Host launcher
// ---------------------------------------------------------------------------
extern "C" void kernel_launch(void* const* d_in, const int* in_sizes, int n_in,
                              void* d_out, int out_size, void* d_ws, size_t ws_size,
                              hipStream_t stream) {
    const float* x  = (const float*)d_in[0];
    const float* Wq = (const float*)d_in[1];
    const float* bq = (const float*)d_in[2];
    const float* Wk = (const float*)d_in[3];
    const float* bk = (const float*)d_in[4];
    const float* Wv = (const float*)d_in[5];
    const float* bv = (const float*)d_in[6];
    const float* Wo = (const float*)d_in[7];
    const float* bo = (const float*)d_in[8];
    float* out = (float*)d_out;

    char* ws = (char*)d_ws;
    const size_t XSZ = (size_t)MROWS * D_MODEL * sizeof(__bf16);   // 32 MB
    const size_t WSZ = (size_t)D_MODEL * D_MODEL * sizeof(__bf16); //  2 MB
    const size_t KVN = (size_t)BSZ * N_HEADS * HEAD_DIM * HEAD_DIM; // 262144

    __bf16* xb   = (__bf16*)(ws);                      // reused as ctx later
    __bf16* wqb  = (__bf16*)(ws + XSZ);
    __bf16* wkb  = (__bf16*)(ws + XSZ + WSZ);
    __bf16* wvb  = (__bf16*)(ws + XSZ + 2 * WSZ);
    __bf16* wob  = (__bf16*)(ws + XSZ + 3 * WSZ);
    __bf16* qb   = (__bf16*)(ws + XSZ + 4 * WSZ);
    __bf16* kb   = (__bf16*)(ws + 2 * XSZ + 4 * WSZ);
    __bf16* vb   = (__bf16*)(ws + 3 * XSZ + 4 * WSZ);
    __bf16* kvT  = (__bf16*)(ws + 4 * XSZ + 4 * WSZ);            // 512 KB
    float*  kvAcc = (float*)(ws + 4 * XSZ + 4 * WSZ + KVN * 2);  // 1 MB
    float*  ksAcc = (float*)(ws + 4 * XSZ + 4 * WSZ + KVN * 2 + KVN * 4);
    __bf16* ctx  = xb;

    // 1) fp32 -> bf16 conversions; zero the fp32 accumulators
    f2bf_kernel<<<2048, 256, 0, stream>>>(x,  xb,  MROWS * D_MODEL);
    f2bf_kernel<<<512,  256, 0, stream>>>(Wq, wqb, D_MODEL * D_MODEL);
    f2bf_kernel<<<512,  256, 0, stream>>>(Wk, wkb, D_MODEL * D_MODEL);
    f2bf_kernel<<<512,  256, 0, stream>>>(Wv, wvb, D_MODEL * D_MODEL);
    f2bf_kernel<<<512,  256, 0, stream>>>(Wo, wob, D_MODEL * D_MODEL);
    zero_kernel<<<256, 256, 0, stream>>>(kvAcc, (int)KVN + BSZ * N_HEADS * HEAD_DIM);

    // 2) Q/K/V projections (WMMA + async LDS staging), per-head layout out
    dim3 ggrid(D_MODEL / 64, MROWS / 256);
    gemm_bf16_kernel<true, false><<<ggrid, 256, 0, stream>>>(xb, wqb, bq, qb, 1);
    gemm_bf16_kernel<true, false><<<ggrid, 256, 0, stream>>>(xb, wkb, bk, kb, 1);
    gemm_bf16_kernel<true, false><<<ggrid, 256, 0, stream>>>(xb, wvb, bv, vb, 0);

    // 3) per-head kv state + k_sum (WMMA, split-N atomics), then bf16 convert
    kv_ksum_kernel<<<dim3(BSZ * N_HEADS, KVSPLIT), 256, 0, stream>>>(kb, vb, kvAcc, ksAcc);
    f2bf_kernel<<<256, 256, 0, stream>>>(kvAcc, kvT, (int)KVN);

    // 4) qkv + normalizer -> ctx [B*N, 1024] bf16
    qkv_norm_kernel<<<dim3(NSEQ / 128, BSZ * N_HEADS), 256, 0, stream>>>(
        qb, kvT, ksAcc, ctx);

    // 5) output projection (WMMA), fp32 result
    gemm_bf16_kernel<false, true><<<ggrid, 256, 0, stream>>>(ctx, wob, bo, out, 0);

    (void)in_sizes; (void)n_in; (void)out_size; (void)ws_size;
}
